// DimeNet_18322330485447
// MI455X (gfx1250) — matrix operand order repaired and checked by reading, not compiled, not run on hardware
//
#include <hip/hip_runtime.h>
#include <hip/hip_bf16.h>

// ---------------------------------------------------------------------------
// DimeNet forward for MI455X (gfx1250, wave32, WMMA).
// All H=128 GEMMs run on v_wmma_f32_16x16x32_f16 (f16 A/B, f32 acc).
// f16 activation copies keep gather tables (x_kj: 38MB) resident in 192MB L2.
// B-operands are staged in LDS in *fragment-major* order (2x ds_load_b128 per
// fragment) and double-buffered in registers so DS latency overlaps WMMA.
// ---------------------------------------------------------------------------

typedef __attribute__((ext_vector_type(16))) _Float16 v16h;
typedef __attribute__((ext_vector_type(8)))  float    v8f;

union Frag16 { v16h h; unsigned u[8]; };

#define NN  20000
#define EE  150000
#define TT  400000
#define GG  16
#define HH  128

__device__ __forceinline__ float silu_f(float x) { return x / (1.f + __expf(-x)); }

// Stage one 128x128 f16 weight chunk into LDS in fragment-major order:
//   Ws[(((ks*8 + ct)*32) + lane)*16 + 2p + e]  holds  W[k][n]
// with k = ks*32 + 16*hi + 2p + e, n = ct*16 + (lane&15), lane = hi*16+(n&15).
// A lane's whole 16-half B fragment is then 32 contiguous bytes.
__device__ __forceinline__ void stage_w_fragmajor(const _Float16* __restrict__ W,
                                                  _Float16* Ws, int tid) {
  for (int c = tid; c < 2048; c += 128) {          // 2048 uint4 = 16384 halves
    int k = c >> 4;                                 // 16 uint4 per 128-wide row
    int s = c & 15;
    uint4 v = *((const uint4*)(W + (long)k * 128) + s);
    const _Float16* hv = (const _Float16*)&v;
    int ks = k >> 5, hi = (k >> 4) & 1, p = (k & 15) >> 1, e = k & 1;
    int n0 = s * 8;
#pragma unroll
    for (int q = 0; q < 8; ++q) {
      int n = n0 + q;
      int lane = hi * 16 + (n & 15);
      Ws[(((ks * 8 + (n >> 4)) * 32) + lane) * 16 + 2 * p + e] = hv[q];
    }
  }
}

// A-fragment (16x32 f16), contiguous k order as in B: lane m=lane&15, hi=lane>>4,
// frag half pair p covers K = kk + 16*hi + 2p.
// Halves p=0..3 and p=4..7 are contiguous -> two ds_load_b128.
__device__ __forceinline__ v16h load_a_frag(const _Float16* As, int rowBase,
                                            int lda, int kk, int lane) {
  Frag16 a;
  int m = lane & 15, hi = lane >> 4;
  const _Float16* ap = As + (rowBase + m) * lda + kk;
#pragma unroll
  for (int p = 0; p < 8; ++p)
    a.u[p] = *(const unsigned*)(ap + 2 * p + 16 * hi);
  return a.h;
}

// B-fragment (32x16 f16): 32 contiguous bytes in fragment-major LDS.
__device__ __forceinline__ v16h load_b_frag(const _Float16* Ws, int ks, int ct,
                                            int lane) {
  Frag16 b;
  const uint4* p = (const uint4*)(Ws + (((ks * 8 + ct) * 32) + lane) * 16);
  uint4 lo = p[0], hi4 = p[1];
  b.u[0] = lo.x;  b.u[1] = lo.y;  b.u[2] = lo.z;  b.u[3] = lo.w;
  b.u[4] = hi4.x; b.u[5] = hi4.y; b.u[6] = hi4.z; b.u[7] = hi4.w;
  return b.h;
}

// ---------------------------------------------------------------------------
// Generic WMMA GEMM: OUT[rows,128] = epilogue(A[rows,K] @ W[K,128]).
// 64 rows per workgroup (4 waves x 16-row M tile), 8 N-tiles per wave.
// CONCAT variant builds A rows as [hn16[i] | hn16[j] | hrbf16[e]] (K=384).
// ---------------------------------------------------------------------------
template <bool BIAS, bool SILU, bool MUL, bool ADD, bool O16, bool O32, bool CONCAT>
__global__ __launch_bounds__(128) void gemm_tiled(
    const _Float16* __restrict__ A16, const _Float16* __restrict__ A16b,
    const int* __restrict__ gi, const int* __restrict__ gj,
    const _Float16* __restrict__ W16, const float* __restrict__ bias,
    const _Float16* __restrict__ mul16, const float* __restrict__ add32,
    _Float16* __restrict__ out16, float* __restrict__ out32, long rows) {
  constexpr int K = CONCAT ? 384 : 128;
  constexpr int SEGS = K / 8;        // uint4 segments per A row
  constexpr int NKC = K / 128;       // K chunks of 128
  extern __shared__ char smemraw[];
  _Float16* As = (_Float16*)smemraw;                       // 64*K halves
  _Float16* Ws = (_Float16*)(smemraw + 64 * K * 2);        // 16384 halves

  const int tid = threadIdx.x;
  const int wave = tid >> 5;
  const int lane = tid & 31;
  const long rowblk = (long)blockIdx.x * 64;

  // ---- stage A tile (64 x K) as f16, zero-padded at the tail ----
  for (int c = tid; c < 64 * SEGS; c += 128) {
    int r = c / SEGS, s = c - r * SEGS;
    long gr = rowblk + r;
    uint4 v = make_uint4(0u, 0u, 0u, 0u);
    if (gr < rows) {
      if (CONCAT) {
        const _Float16* src;
        if (s < 16)      src = A16 + (long)gi[gr] * 128 + s * 8;
        else if (s < 32) src = A16 + (long)gj[gr] * 128 + (s - 16) * 8;
        else             src = A16b + gr * 128 + (s - 32) * 8;
        v = *(const uint4*)src;
      } else {
        v = *(const uint4*)(A16 + gr * 128 + s * 8);
      }
    }
    *(uint4*)(As + r * K + s * 8) = v;
  }

  v8f acc[8];
#pragma unroll
  for (int ct = 0; ct < 8; ++ct) {
    float b = BIAS ? bias[ct * 16 + (lane & 15)] : 0.f;
#pragma unroll
    for (int v = 0; v < 8; ++v) acc[ct][v] = b;
  }

  for (int kc = 0; kc < NKC; ++kc) {
    __syncthreads();
    stage_w_fragmajor(W16 + (long)kc * 128 * 128, Ws, tid);
    __syncthreads();
    // hoist the 4 A fragments for this K chunk
    v16h a4[4];
#pragma unroll
    for (int ks = 0; ks < 4; ++ks)
      a4[ks] = load_a_frag(As, wave * 16, K, kc * 128 + ks * 32, lane);
    // double-buffered B fragments: load (it+1) before WMMA(it)
    v16h bcur = load_b_frag(Ws, 0, 0, lane);
#pragma unroll
    for (int it = 0; it < 32; ++it) {
      const int ks = it >> 3, ct = it & 7;
      v16h bnext = bcur;
      if (it < 31) bnext = load_b_frag(Ws, (it + 1) >> 3, (it + 1) & 7, lane);
      acc[ct] = __builtin_amdgcn_wmma_f32_16x16x32_f16(
          false, a4[ks], false, bcur, (short)0, acc[ct], false, false);
      bcur = bnext;
    }
  }

  // ---- fused epilogue: bias (pre-added) -> SiLU -> *mul -> +add -> stores
  const int m = lane & 15, hi = lane >> 4;
#pragma unroll
  for (int ct = 0; ct < 8; ++ct) {
#pragma unroll
    for (int v = 0; v < 8; ++v) {
      long gr = rowblk + wave * 16 + v + 8 * hi;
      if (gr >= rows) continue;
      int n = ct * 16 + m;
      float x = acc[ct][v];
      if (SILU) x = silu_f(x);
      if (MUL)  x *= (float)mul16[gr * 128 + n];
      if (ADD)  x += add32[gr * 128 + n];
      if (O32) out32[gr * 128 + n] = x;
      if (O16) out16[gr * 128 + n] = (_Float16)x;
    }
  }
}

// ---------------------------------------------------------------------------
// Triplet bilinear: for each triplet t, msg[t,:] = sum_jb sbf_b[t,jb] *
// (x_kj[idx_kj[t],:] @ Wt[jb]) ; atomically scattered into agg[idx_ji[t],:].
// 64 triplets per workgroup; Wt[jb] (128x128 f16, pre-transposed so it is the
// direct WMMA B operand) cycled through LDS with prefetch of the next slice.
// A fragments hoisted (invariant over jb); B fragments double-buffered.
// ---------------------------------------------------------------------------
__global__ __launch_bounds__(128) void bilinear_kernel(
    const _Float16* __restrict__ xkj16, const int* __restrict__ idx_kj,
    const int* __restrict__ idx_ji, const float* __restrict__ sbfb,
    const _Float16* __restrict__ Wt16, float* __restrict__ agg) {
  __shared__ _Float16 As[64 * 128];
  __shared__ _Float16 Ws[16384];
  __shared__ float sbL[64 * 8];
  __shared__ int ejL[64];

  const int tid = threadIdx.x;
  const int wave = tid >> 5;
  const int lane = tid & 31;
  const long rowblk = (long)blockIdx.x * 64;

  // gather A tile (L2-resident f16 table)
  for (int c = tid; c < 64 * 16; c += 128) {
    int r = c >> 4, s = c & 15;
    long src = (long)idx_kj[rowblk + r] * 128;
    *(uint4*)(As + r * 128 + s * 8) = *(const uint4*)(xkj16 + src + s * 8);
  }
  for (int c = tid; c < 512; c += 128) sbL[c] = sbfb[rowblk * 8 + c];
  if (tid < 64) ejL[tid] = idx_ji[rowblk + tid];
  __syncthreads();

  // hoist the 4 A fragments (x_kj tile is invariant over jb)
  v16h a4[4];
#pragma unroll
  for (int ks = 0; ks < 4; ++ks)
    a4[ks] = load_a_frag(As, wave * 16, 128, ks * 32, lane);

  v8f macc[8];
#pragma unroll
  for (int ct = 0; ct < 8; ++ct)
#pragma unroll
    for (int v = 0; v < 8; ++v) macc[ct][v] = 0.f;

  const int m = lane & 15, hi = lane >> 4;

  for (int jb = 0; jb < 8; ++jb) {
    __syncthreads();
    stage_w_fragmajor(Wt16 + (long)jb * 16384, Ws, tid);
    if (jb < 7) __builtin_prefetch(Wt16 + (long)(jb + 1) * 16384, 0, 0);
    __syncthreads();

    float sv[8];
#pragma unroll
    for (int v = 0; v < 8; ++v)
      sv[v] = sbL[(wave * 16 + v + 8 * hi) * 8 + jb];

    // double-buffered B fragments across the flattened (ct,ks) iteration
    v16h bcur = load_b_frag(Ws, 0, 0, lane);
#pragma unroll
    for (int ct = 0; ct < 8; ++ct) {
      v8f acc;
#pragma unroll
      for (int v = 0; v < 8; ++v) acc[v] = 0.f;
#pragma unroll
      for (int ks = 0; ks < 4; ++ks) {
        v16h bnext = bcur;
        const int nit = ct * 4 + ks + 1;
        if (nit < 32) bnext = load_b_frag(Ws, nit & 3, nit >> 2, lane);
        acc = __builtin_amdgcn_wmma_f32_16x16x32_f16(
            false, a4[ks], false, bcur, (short)0, acc, false, false);
        bcur = bnext;
      }
#pragma unroll
      for (int v = 0; v < 8; ++v) macc[ct][v] += sv[v] * acc[v];
    }
  }

  // scatter (segment_sum over idx_ji) with global f32 atomics
#pragma unroll
  for (int ct = 0; ct < 8; ++ct) {
#pragma unroll
    for (int v = 0; v < 8; ++v) {
      int row = wave * 16 + v + 8 * hi;
      long e = (long)ejL[row];
      atomicAdd(&agg[e * 128 + ct * 16 + m], macc[ct][v]);
    }
  }
}

// ------------------------- small VALU kernels ------------------------------

__global__ __launch_bounds__(128) void k_hn(const float* __restrict__ x,
                                            const float* __restrict__ W,
                                            const float* __restrict__ b,
                                            _Float16* __restrict__ hn16) {
  __shared__ float Wl[21 * 128];
  __shared__ float xr[21];
  int tid = threadIdx.x;
  long n = blockIdx.x;
  for (int c = tid; c < 21 * 128; c += 128) Wl[c] = W[c];
  if (tid < 21) xr[tid] = x[n * 21 + tid];
  __syncthreads();
  float s = b[tid];
#pragma unroll
  for (int a = 0; a < 21; ++a) s += xr[a] * Wl[a * 128 + tid];
  hn16[n * 128 + tid] = (_Float16)s;
}

template <bool BIAS, bool SILU>
__global__ __launch_bounds__(128) void k_rbf(const float* __restrict__ rbf,
                                             const float* __restrict__ W,
                                             const float* __restrict__ bias,
                                             _Float16* __restrict__ out16,
                                             long nrows) {
  int h = threadIdx.x;
  float w[6];
#pragma unroll
  for (int r = 0; r < 6; ++r) w[r] = W[r * 128 + h];
  float b = BIAS ? bias[h] : 0.f;
  long e0 = (long)blockIdx.x * 32;
  for (int k = 0; k < 32; ++k) {
    long e = e0 + k;
    if (e >= nrows) return;
    float s = b;
#pragma unroll
    for (int r = 0; r < 6; ++r) s += rbf[e * 6 + r] * w[r];
    if (SILU) s = silu_f(s);
    out16[e * 128 + h] = (_Float16)s;
  }
}

__global__ __launch_bounds__(256) void k_sbf(const float* __restrict__ sbf,
                                             const float* __restrict__ W,
                                             float* __restrict__ out, long T) {
  __shared__ float Wl[42 * 8];
  __shared__ float Sl[32 * 42];
  int tid = threadIdx.x;
  long t0 = (long)blockIdx.x * 32;
  for (int c = tid; c < 42 * 8; c += 256) Wl[c] = W[c];
  for (int c = tid; c < 32 * 42; c += 256) {
    long t = t0 + c / 42;
    Sl[c] = (t < T) ? sbf[t * 42 + (c % 42)] : 0.f;
  }
  __syncthreads();
  int lt = tid >> 3, jb = tid & 7;
  long t = t0 + lt;
  if (t < T) {
    float s = 0.f;
#pragma unroll
    for (int k = 0; k < 42; ++k) s += Sl[lt * 42 + k] * Wl[k * 8 + jb];
    out[t * 8 + jb] = s;
  }
}

__global__ void k_cvt(const float* __restrict__ src, _Float16* __restrict__ dst,
                      long n) {
  long i = (long)blockIdx.x * blockDim.x + threadIdx.x;
  if (i < n) dst[i] = (_Float16)src[i];
}

// bilinear W[i,jb,l] (128,8,128) -> Wt[jb][l][i] f16 (direct B operand)
__global__ void k_cvt_bil(const float* __restrict__ W,
                          _Float16* __restrict__ Wt) {
  long t = (long)blockIdx.x * blockDim.x + threadIdx.x;
  if (t >= 8L * 128 * 128) return;
  int jb = (int)(t >> 14);
  int l = (int)((t >> 7) & 127);
  int i = (int)(t & 127);
  Wt[t] = (_Float16)W[(long)i * 1024 + jb * 128 + l];
}

__global__ void k_zero(float* __restrict__ p, long n) {
  long i = (long)blockIdx.x * blockDim.x + threadIdx.x;
  if (i < n) p[i] = 0.f;
}

// g[i[e],:] += rbf_h[e,:] * m[e,:]
__global__ __launch_bounds__(128) void k_scale_scatter(
    const _Float16* __restrict__ rbfh16, const float* __restrict__ m32,
    const int* __restrict__ iidx, float* __restrict__ g, long E) {
  int h = threadIdx.x;
  long e0 = (long)blockIdx.x * 16;
  for (int k = 0; k < 16; ++k) {
    long e = e0 + k;
    if (e >= E) return;
    float v = (float)rbfh16[e * 128 + h] * m32[e * 128 + h];
    atomicAdd(&g[(long)iidx[e] * 128 + h], v);
  }
}

__global__ __launch_bounds__(128) void k_readout(const float* __restrict__ P,
                                                 const int* __restrict__ batch,
                                                 float* __restrict__ out,
                                                 float* __restrict__ cnt) {
  long n = blockIdx.x;
  int h = threadIdx.x;
  int g = batch[n];
  atomicAdd(&out[(long)g * 128 + h], P[n * 128 + h]);
  if (h == 0) atomicAdd(&cnt[g], 1.f);
}

__global__ __launch_bounds__(128) void k_div(float* __restrict__ out,
                                             const float* __restrict__ cnt) {
  int g = blockIdx.x, h = threadIdx.x;
  out[(long)g * 128 + h] /= fmaxf(cnt[g], 1.f);
}

// ---------------------------------------------------------------------------
// Host launcher.
// Param leaves follow jax pytree order (sorted dict keys):
//  d_in[0..7]  = x, rbf, sbf, i, j, idx_kj, idx_ji, batch
//  d_in[8..70] = blocks[0..2] x { W, after[0].l1.{b,w}, after[0].l2.{b,w},
//                 after[1].l1.{b,w}, after[1].l2.{b,w}, before[0].l1.{b,w},
//                 before[0].l2.{b,w}, lin.{b,w}, lin_ji.{b,w}, lin_kj.{b,w},
//                 lin_rbf.w, lin_sbf.w }                      (21 per block)
//  d_in[71..76]= emb.{b,w}, lin.{b,w}, rbf.{b,w}
//  d_in[77..108]= out[0..3] x { lin.w, lin_rbf.w, lins[0..2].{b,w} }
// ---------------------------------------------------------------------------
extern "C" void kernel_launch(void* const* d_in, const int* in_sizes, int n_in,
                              void* d_out, int out_size, void* d_ws,
                              size_t ws_size, hipStream_t stream) {
  if (n_in < 109) return;
  const float* x      = (const float*)d_in[0];
  const float* rbf    = (const float*)d_in[1];
  const float* sbf    = (const float*)d_in[2];
  const int* ii       = (const int*)d_in[3];
  const int* jjj      = (const int*)d_in[4];
  const int* idx_kj   = (const int*)d_in[5];
  const int* idx_ji   = (const int*)d_in[6];
  const int* batch    = (const int*)d_in[7];
  auto prm = [&](int k) -> const float* { return (const float*)d_in[k]; };

  // ---- workspace bump allocator (256B aligned) ----
  char* ws = (char*)d_ws;
  size_t off = 0;
  auto alloc = [&](size_t bytes) -> char* {
    off = (off + 255) & ~(size_t)255;
    char* p = ws + off;
    off += bytes;
    return p;
  };

  _Float16* hn16   = (_Float16*)alloc((size_t)NN * HH * 2);
  _Float16* scr16  = (_Float16*)alloc((size_t)EE * HH * 2); // hrbf16 / xkj16 / t16
  float*    m32    = (float*)alloc((size_t)EE * HH * 4);
  _Float16* m16    = (_Float16*)alloc((size_t)EE * HH * 2);
  float*    h32    = (float*)alloc((size_t)EE * HH * 4);
  _Float16* h16    = (_Float16*)alloc((size_t)EE * HH * 2);
  _Float16* rbfh16 = (_Float16*)alloc((size_t)EE * HH * 2);
  float*    sbfb32 = (float*)alloc((size_t)TT * 8 * 4);
  float*    g32    = (float*)alloc((size_t)NN * HH * 4);
  _Float16* g16    = (_Float16*)alloc((size_t)NN * HH * 2);
  float*    P32    = (float*)alloc((size_t)NN * HH * 4);
  float*    cnt    = (float*)alloc(64);

  auto cvt = [&](const float* s, _Float16* d, long n) {
    k_cvt<<<(unsigned)((n + 255) / 256), 256, 0, stream>>>(s, d, n);
  };
  auto mk16 = [&](const float* s, long n) -> _Float16* {
    _Float16* d = (_Float16*)alloc((size_t)n * 2);
    cvt(s, d, n);
    return d;
  };

  // ---- pre-convert WMMA weights to f16 ----
  _Float16* w_emb_lin = mk16(prm(74), 384 * 128);
  _Float16 *w_kj[3], *w_ji[3], *w_lin[3], *w_bl1[3], *w_bl2[3];
  _Float16 *w_al1[3][2], *w_al2[3][2], *w_bil[3];
  for (int b = 0; b < 3; ++b) {
    int o = 8 + 21 * b;
    w_bil[b] = (_Float16*)alloc((size_t)8 * 128 * 128 * 2);
    k_cvt_bil<<<(8 * 128 * 128 + 255) / 256, 256, 0, stream>>>(prm(o + 0), w_bil[b]);
    w_al1[b][0] = mk16(prm(o + 2), 16384);
    w_al2[b][0] = mk16(prm(o + 4), 16384);
    w_al1[b][1] = mk16(prm(o + 6), 16384);
    w_al2[b][1] = mk16(prm(o + 8), 16384);
    w_bl1[b]    = mk16(prm(o + 10), 16384);
    w_bl2[b]    = mk16(prm(o + 12), 16384);
    w_lin[b]    = mk16(prm(o + 14), 16384);
    w_ji[b]     = mk16(prm(o + 16), 16384);
    w_kj[b]     = mk16(prm(o + 18), 16384);
  }
  _Float16 *w_out_fin[4], *w_out_l[4][3];
  for (int k = 0; k < 4; ++k) {
    int o = 77 + 8 * k;
    w_out_fin[k] = mk16(prm(o + 0), 16384);
    w_out_l[k][0] = mk16(prm(o + 3), 16384);
    w_out_l[k][1] = mk16(prm(o + 5), 16384);
    w_out_l[k][2] = mk16(prm(o + 7), 16384);
  }

  const unsigned gE64 = (EE + 63) / 64, gN64 = (NN + 63) / 64;
  const unsigned gE32 = (EE + 31) / 32, gE16 = (EE + 15) / 16;
  const unsigned gT32 = (TT + 31) / 32, gT64 = TT / 64;
  const size_t SH128 = 64 * 128 * 2 + 32768;   // gemm shared (K=128)
  const size_t SH384 = 64 * 384 * 2 + 32768;   // concat gemm shared (K=384)

  // ---- embedding ----
  k_hn<<<NN, 128, 0, stream>>>(x, prm(72), prm(71), hn16);
  k_rbf<true, true><<<gE32, 128, 0, stream>>>(rbf, prm(76), prm(75), scr16, EE);
  // m = silu([hn_i|hn_j|h_rbf] @ W + b)  -> m16 + m32
  gemm_tiled<true, true, false, false, true, true, true>
      <<<gE64, 128, SH384, stream>>>(hn16, scr16, ii, jjj, w_emb_lin, prm(73),
                                     nullptr, nullptr, m16, m32, EE);

  // ---- output block (shared by all 4 uses) ----
  auto out_block = [&](int k, bool first) {
    int o = 77 + 8 * k;
    k_rbf<false, false><<<gE32, 128, 0, stream>>>(rbf, prm(o + 1), nullptr,
                                                  rbfh16, EE);
    k_zero<<<(NN * HH + 255) / 256, 256, 0, stream>>>(g32, (long)NN * HH);
    k_scale_scatter<<<gE16, 128, 0, stream>>>(rbfh16, m32, ii, g32, EE);
    cvt(g32, g16, (long)NN * HH);
    for (int l = 0; l < 3; ++l)
      gemm_tiled<true, true, false, false, true, false, false>
          <<<gN64, 128, SH128, stream>>>(g16, nullptr, nullptr, nullptr,
                                         w_out_l[k][l], prm(o + 2 + 2 * l),
                                         nullptr, nullptr, g16, nullptr, NN);
    if (first)
      gemm_tiled<false, false, false, false, false, true, false>
          <<<gN64, 128, SH128, stream>>>(g16, nullptr, nullptr, nullptr,
                                         w_out_fin[k], nullptr, nullptr,
                                         nullptr, nullptr, P32, NN);
    else
      gemm_tiled<false, false, false, true, false, true, false>
          <<<gN64, 128, SH128, stream>>>(g16, nullptr, nullptr, nullptr,
                                         w_out_fin[k], nullptr, nullptr, P32,
                                         nullptr, P32, NN);
  };

  out_block(0, true);

  // ---- interaction blocks ----
  for (int b = 0; b < 3; ++b) {
    int o = 8 + 21 * b;
    k_rbf<false, false><<<gE32, 128, 0, stream>>>(rbf, prm(o + 19), nullptr,
                                                  rbfh16, EE);
    k_sbf<<<gT32, 256, 0, stream>>>(sbf, prm(o + 20), sbfb32, TT);
    // x_kj = silu(m @ Wkj + b) * rbf_h   -> f16 gather table
    gemm_tiled<true, true, true, false, true, false, false>
        <<<gE64, 128, SH128, stream>>>(m16, nullptr, nullptr, nullptr, w_kj[b],
                                       prm(o + 17), rbfh16, nullptr, scr16,
                                       nullptr, EE);
    // h = x_ji = silu(m @ Wji + b)       -> f32 accumulator
    gemm_tiled<true, true, false, false, false, true, false>
        <<<gE64, 128, SH128, stream>>>(m16, nullptr, nullptr, nullptr, w_ji[b],
                                       prm(o + 15), nullptr, nullptr, nullptr,
                                       h32, EE);
    // h += segment_sum(bilinear msg, idx_ji)
    bilinear_kernel<<<gT64, 128, 0, stream>>>(scr16, idx_kj, idx_ji, sbfb32,
                                              w_bil[b], h32);
    cvt(h32, h16, (long)EE * HH);
    // before residual
    gemm_tiled<true, true, false, false, true, false, false>
        <<<gE64, 128, SH128, stream>>>(h16, nullptr, nullptr, nullptr,
                                       w_bl1[b], prm(o + 9), nullptr, nullptr,
                                       scr16, nullptr, EE);
    gemm_tiled<true, true, false, true, true, true, false>
        <<<gE64, 128, SH128, stream>>>(scr16, nullptr, nullptr, nullptr,
                                       w_bl2[b], prm(o + 11), nullptr, h32,
                                       h16, h32, EE);
    // h = silu(h @ Wlin + b) + m
    gemm_tiled<true, true, false, true, true, true, false>
        <<<gE64, 128, SH128, stream>>>(h16, nullptr, nullptr, nullptr,
                                       w_lin[b], prm(o + 13), nullptr, m32,
                                       h16, h32, EE);
    // after residuals
    for (int r = 0; r < 2; ++r) {
      gemm_tiled<true, true, false, false, true, false, false>
          <<<gE64, 128, SH128, stream>>>(h16, nullptr, nullptr, nullptr,
                                         w_al1[b][r], prm(o + 1 + 4 * r),
                                         nullptr, nullptr, scr16, nullptr, EE);
      gemm_tiled<true, true, false, true, true, true, false>
          <<<gE64, 128, SH128, stream>>>(scr16, nullptr, nullptr, nullptr,
                                         w_al2[b][r], prm(o + 3 + 4 * r),
                                         nullptr, h32, h16, h32, EE);
    }
    // h becomes the new m
    { float* t = m32; m32 = h32; h32 = t; }
    { _Float16* t = m16; m16 = h16; h16 = t; }
    out_block(b + 1, false);
  }

  // ---- mean readout over graphs ----
  float* out = (float*)d_out;
  k_zero<<<(GG * HH + 255) / 256, 256, 0, stream>>>(out, (long)GG * HH);
  k_zero<<<1, 64, 0, stream>>>(cnt, 16);
  k_readout<<<NN, 128, 0, stream>>>(P32, batch, out, cnt);
  k_div<<<GG, 128, 0, stream>>>(out, cnt);
}
